// TransformerBlock_80513456931522
// MI455X (gfx1250) — matrix operand order repaired and checked by reading, not compiled, not run on hardware
//
#include <hip/hip_runtime.h>
#include <hip/hip_fp16.h>

typedef __attribute__((ext_vector_type(16))) _Float16 v16h;
typedef __attribute__((ext_vector_type(8)))  _Float16 v8h;
typedef __attribute__((ext_vector_type(4)))  _Float16 v4h;
typedef __attribute__((ext_vector_type(8)))  float    v8f;

#define BB 2
#define NN 4096
#define KK 16
#define HH 64
#define PTS (BB*NN)

// Explicit LDS fence (CDNA5 split counter): order wave-internal LDS
// staging-buffer reuse (RAW/WAR across the C-layout <-> A-layout transposes).
#define DS_FENCE() asm volatile("s_wait_dscnt 0" ::: "memory")

__device__ __forceinline__ v8f wmma_f16(v16h a, v16h b, v8f c) {
  // emits v_wmma_f32_16x16x32_f16
  return __builtin_amdgcn_wmma_f32_16x16x32_f16(false, a, false, b, (short)0, c,
                                                false, false);
}

// A-matrix fragment (16x32 f16) from a row-major [16][64] f16 LDS buffer.
// ISA layout: per lane the 16 halves are TWO contiguous 8-half runs:
//   k = kb..kb+7  and  k = kb+16..kb+23,  kb = ks*32 + (lane>=16)*8.
// -> 2x ds_load_b128 per fragment.
__device__ __forceinline__ v16h load_a_frag(const _Float16* s, int lane, int ks) {
  int row = lane & 15;
  int kb = ks * 32 + ((lane >> 4) & 1) * 8;
  v8h lo = *(const v8h*)(s + row * HH + kb);
  v8h hi = *(const v8h*)(s + row * HH + kb + 16);
  return __builtin_shufflevector(lo, hi, 0, 1, 2, 3, 4, 5, 6, 7,
                                 8, 9, 10, 11, 12, 13, 14, 15);
}

// B-matrix fragment (32x16 f16) from LDS weights pre-swizzled fragment-major:
// v16h slot [(ks*4+nt)*32 + lane] holds that lane's 16 halves contiguously.
// -> one aligned v16h load (2x ds_load_b128).
__device__ __forceinline__ v16h load_b_frag(const v16h* w, int lane, int ks,
                                            int nt) {
  return w[((ks << 2) + nt) * 32 + lane];
}

// Block-cooperative: convert a row-major [64][64] f32 weight matrix into the
// fragment-major f16 LDS image expected by load_b_frag.
//   d = ((ks*4+nt)*32 + lane)*16 + i  maps to  W[k][col],
//   k = ks*32 + (lane>=16)*8 + (i&7) + (i>>3)*16,  col = nt*16 + (lane&15).
__device__ __forceinline__ void stage_w_swizzled(v16h* dst,
                                                 const float* __restrict__ W,
                                                 int tid) {
  _Float16* out = (_Float16*)dst;
  for (int d = tid; d < HH * HH; d += 256) {
    int i = d & 15;
    int ln = (d >> 4) & 31;
    int tile = d >> 9;                // ks*4+nt
    int ks = tile >> 2, nt = tile & 3;
    int k = ks * 32 + ((ln >> 4) & 1) * 8 + (i & 7) + (i >> 3) * 16;
    int col = nt * 16 + (ln & 15);
    out[d] = (_Float16)W[k * HH + col];
  }
}

// Gather 16 rows of 64 f32 (row ids from sidx) into a row-major f16 staging
// buffer. Each lane moves half a row: 8x global_load_b128 + 8x ds_store_b64.
__device__ __forceinline__ void stage_gather(_Float16* st,
                                             const float* __restrict__ base,
                                             const int* sidx, int lane) {
  int r = lane >> 1;
  int c0 = (lane & 1) * 32;
  const float4* src = (const float4*)(base + (size_t)sidx[r] * HH + c0);
#pragma unroll
  for (int i = 0; i < 8; ++i) {
    float4 v = src[i];
    v4h t;
    t[0] = (_Float16)v.x; t[1] = (_Float16)v.y;
    t[2] = (_Float16)v.z; t[3] = (_Float16)v.w;
    *(v4h*)(st + r * HH + c0 + i * 4) = t;
  }
}

// ---------------------------------------------------------------------------
// Kernel A: x = feat@Wk+bk; q = x@Wq; xk = x@Wkk; xv = x@Wv  (all WMMA f16)
// One wave (wave32) owns one 16-row tile; 8 waves / block.
// ---------------------------------------------------------------------------
__global__ __launch_bounds__(256) void qkv_kernel(
    const float* __restrict__ features, const float* __restrict__ Wk,
    const float* __restrict__ bk, const float* __restrict__ Wq,
    const float* __restrict__ Wkk, const float* __restrict__ Wv,
    float* __restrict__ qo, float* __restrict__ xko, float* __restrict__ xvo) {
  __shared__ v16h swk[256], swq[256], swkk[256], swv[256];
  __shared__ float sbk[HH];
  __shared__ _Float16 stg[8][16 * HH];

  int tid = threadIdx.x;
  stage_w_swizzled(swk, Wk, tid);
  stage_w_swizzled(swq, Wq, tid);
  stage_w_swizzled(swkk, Wkk, tid);
  stage_w_swizzled(swv, Wv, tid);
  if (tid < HH) sbk[tid] = bk[tid];
  __syncthreads();

  int lane = tid & 31, wid = tid >> 5;
  int m0 = (blockIdx.x * 8 + wid) * 16;
  _Float16* st = stg[wid];
  int l15 = lane & 15, hi = lane >> 4;

  // stage this tile's features as f16 (coalesced float4 loads)
  {
    int r = lane >> 1, c0 = (lane & 1) * 32;
    const float4* src = (const float4*)(features + (size_t)(m0 + r) * HH + c0);
#pragma unroll
    for (int i = 0; i < 8; ++i) {
      float4 v = src[i];
      v4h t;
      t[0] = (_Float16)v.x; t[1] = (_Float16)v.y;
      t[2] = (_Float16)v.z; t[3] = (_Float16)v.w;
      *(v4h*)(st + r * HH + c0 + i * 4) = t;
    }
  }
  DS_FENCE();
  v16h af0 = load_a_frag(st, lane, 0), af1 = load_a_frag(st, lane, 1);

  // x = features @ Wk + bk
  v8f xacc[4];
#pragma unroll
  for (int nt = 0; nt < 4; ++nt) {
    v8f acc;
    float bias = sbk[nt * 16 + l15];
#pragma unroll
    for (int r = 0; r < 8; ++r) acc[r] = bias;
    acc = wmma_f16(af0, load_b_frag(swk, lane, 0, nt), acc);
    acc = wmma_f16(af1, load_b_frag(swk, lane, 1, nt), acc);
    xacc[nt] = acc;
  }
  DS_FENCE();
  // transpose C-layout -> A-layout through LDS
#pragma unroll
  for (int nt = 0; nt < 4; ++nt)
#pragma unroll
    for (int r = 0; r < 8; ++r)
      st[(r + 8 * hi) * HH + nt * 16 + l15] = (_Float16)xacc[nt][r];
  DS_FENCE();
  v16h ax0 = load_a_frag(st, lane, 0), ax1 = load_a_frag(st, lane, 1);

  const v16h* ws3[3] = {swq, swkk, swv};
  float* outs[3] = {qo, xko, xvo};
#pragma unroll
  for (int m = 0; m < 3; ++m) {
#pragma unroll
    for (int nt = 0; nt < 4; ++nt) {
      v8f acc = {};
      acc = wmma_f16(ax0, load_b_frag(ws3[m], lane, 0, nt), acc);
      acc = wmma_f16(ax1, load_b_frag(ws3[m], lane, 1, nt), acc);
#pragma unroll
      for (int r = 0; r < 8; ++r)
        outs[m][(size_t)(m0 + r + 8 * hi) * HH + nt * 16 + l15] = acc[r];
    }
  }
}

// ---------------------------------------------------------------------------
// Kernel B: brute-force KNN (top-16 smallest d2, self included).
// One thread per query; candidates tiled through LDS; register insertion sort.
// ---------------------------------------------------------------------------
__global__ __launch_bounds__(256) void knn_kernel(const float* __restrict__ xyz,
                                                  int* __restrict__ idx_out) {
  __shared__ float sx[256], sy[256], sz[256];
  int gq = blockIdx.x * 256 + threadIdx.x;
  int b = gq >> 12;
  int n = gq & (NN - 1);
  const float* base = xyz + (size_t)b * NN * 3;
  float qx = base[n * 3 + 0], qy = base[n * 3 + 1], qz = base[n * 3 + 2];
  float bd[KK];
  int bi[KK];
#pragma unroll
  for (int i = 0; i < KK; ++i) { bd[i] = 3.4e38f; bi[i] = 0; }

  for (int t0 = 0; t0 < NN; t0 += 256) {
    int j = t0 + threadIdx.x;
    sx[threadIdx.x] = base[j * 3 + 0];
    sy[threadIdx.x] = base[j * 3 + 1];
    sz[threadIdx.x] = base[j * 3 + 2];
    if (t0 + 256 < NN)  // speculative prefetch of next tile -> global_prefetch_b8
      __builtin_prefetch(base + (size_t)(t0 + 256 + threadIdx.x) * 3, 0, 0);
    __syncthreads();
    for (int c = 0; c < 256; ++c) {
      float dx = qx - sx[c], dy = qy - sy[c], dz = qz - sz[c];
      float d = fmaf(dx, dx, fmaf(dy, dy, dz * dz));
      if (d < bd[KK - 1]) {
        float cd = d;
        int ci = t0 + c;
#pragma unroll
        for (int i = 0; i < KK; ++i) {
          if (cd < bd[i]) {
            float td = bd[i]; int ti = bi[i];
            bd[i] = cd; bi[i] = ci;
            cd = td; ci = ti;
          }
        }
      }
    }
    __syncthreads();
  }
#pragma unroll
  for (int i = 0; i < KK; ++i) idx_out[(size_t)gq * KK + i] = bi[i];
}

// ---------------------------------------------------------------------------
// Kernel C: per-point attention. One wave per point (M=16 = K neighbors).
// pos_enc MLP + attention MLP on WMMA; softmax over K via 8-reg reduce +
// __shfl_xor(16) half-wave swap; final matvec + residual.
// ---------------------------------------------------------------------------
__global__ __launch_bounds__(256) void attn_kernel(
    const float* __restrict__ xyz, const float* __restrict__ features,
    const float* __restrict__ pe_w1, const float* __restrict__ pe_b1,
    const float* __restrict__ pe_w2, const float* __restrict__ pe_b2,
    const float* __restrict__ at_w1, const float* __restrict__ at_b1,
    const float* __restrict__ at_w2, const float* __restrict__ at_b2,
    const float* __restrict__ Wa, const float* __restrict__ ba,
    const float* __restrict__ qb, const float* __restrict__ xkb,
    const float* __restrict__ xvb, const int* __restrict__ knn,
    float* __restrict__ out_res, float* __restrict__ out_attn) {
  __shared__ v16h s_pew2[256], s_atw1[256], s_atw2[256];
  __shared__ float s_Wa[HH * HH];
  __shared__ float s_pew1[3 * HH];
  __shared__ float s_peb1[HH], s_peb2[HH], s_atb1[HH], s_atb2[HH], s_ba[HH];
  __shared__ _Float16 stg[8][16 * HH];
  __shared__ int sidxs[8][KK];
  __shared__ float sres[8][HH];

  int tid = threadIdx.x;
  stage_w_swizzled(s_pew2, pe_w2, tid);
  stage_w_swizzled(s_atw1, at_w1, tid);
  stage_w_swizzled(s_atw2, at_w2, tid);
  for (int i = tid; i < HH * HH; i += 256) s_Wa[i] = Wa[i];
  if (tid < 3 * HH) s_pew1[tid] = pe_w1[tid];
  if (tid < HH) {
    s_peb1[tid] = pe_b1[tid]; s_peb2[tid] = pe_b2[tid];
    s_atb1[tid] = at_b1[tid]; s_atb2[tid] = at_b2[tid];
    s_ba[tid] = ba[tid];
  }
  __syncthreads();

  int lane = tid & 31, wid = tid >> 5;
  int p = blockIdx.x * 8 + wid;
  int b = p >> 12, n = p & (NN - 1);
  _Float16* st = stg[wid];
  int* sidx = sidxs[wid];
  int l15 = lane & 15, hi = lane >> 4;

  if (lane < KK) sidx[lane] = knn[(size_t)p * KK + lane];
  DS_FENCE();

  // ---- h1 = relu(rel @ pe_w1 + pe_b1), staged f16 [16][64] (v8h stores)
  {
    int k = lane >> 1;
    int nb = sidx[k];
    const float* cb = xyz + ((size_t)b * NN + n) * 3;
    const float* kb = xyz + ((size_t)b * NN + nb) * 3;
    float rx = cb[0] - kb[0], ry = cb[1] - kb[1], rz = cb[2] - kb[2];
    int h0 = (lane & 1) * 32;
#pragma unroll
    for (int g = 0; g < 4; ++g) {
      v8h tmp;
#pragma unroll
      for (int i = 0; i < 8; ++i) {
        int h = h0 + g * 8 + i;
        float v = fmaf(rx, s_pew1[h],
                  fmaf(ry, s_pew1[HH + h],
                  fmaf(rz, s_pew1[2 * HH + h], s_peb1[h])));
        tmp[i] = (_Float16)fmaxf(v, 0.f);
      }
      *(v8h*)(st + k * HH + h0 + g * 8) = tmp;
    }
  }
  DS_FENCE();

  // ---- pos_enc = h1 @ pe_w2 + pe_b2  (kept in C-layout registers)
  v16h fa0 = load_a_frag(st, lane, 0), fa1 = load_a_frag(st, lane, 1);
  v8f pe[4];
#pragma unroll
  for (int nt = 0; nt < 4; ++nt) {
    v8f acc;
    float bias = s_peb2[nt * 16 + l15];
#pragma unroll
    for (int r = 0; r < 8; ++r) acc[r] = bias;
    acc = wmma_f16(fa0, load_b_frag(s_pew2, lane, 0, nt), acc);
    acc = wmma_f16(fa1, load_b_frag(s_pew2, lane, 1, nt), acc);
    pe[nt] = acc;
  }

  // ---- gather k_ rows, build a = q - k_ + pos_enc in staging
  DS_FENCE();
  stage_gather(st, xkb + (size_t)b * NN * HH, sidx, lane);
  DS_FENCE();

  float qv[4];
#pragma unroll
  for (int t = 0; t < 4; ++t) qv[t] = qb[(size_t)p * HH + t * 16 + l15];

#pragma unroll
  for (int t = 0; t < 4; ++t) {
    int h = t * 16 + l15;
#pragma unroll
    for (int r = 0; r < 8; ++r) {
      int k = r + 8 * hi;
      float av = qv[t] - (float)st[k * HH + h] + pe[t][r];
      st[k * HH + h] = (_Float16)av;
    }
  }
  DS_FENCE();

  // ---- hidden = relu(a @ at_w1 + at_b1)
  fa0 = load_a_frag(st, lane, 0); fa1 = load_a_frag(st, lane, 1);
  v8f hf[4];
#pragma unroll
  for (int nt = 0; nt < 4; ++nt) {
    v8f acc;
    float bias = s_atb1[nt * 16 + l15];
#pragma unroll
    for (int r = 0; r < 8; ++r) acc[r] = bias;
    acc = wmma_f16(fa0, load_b_frag(s_atw1, lane, 0, nt), acc);
    acc = wmma_f16(fa1, load_b_frag(s_atw1, lane, 1, nt), acc);
    hf[nt] = acc;
  }
  DS_FENCE();
#pragma unroll
  for (int t = 0; t < 4; ++t)
#pragma unroll
    for (int r = 0; r < 8; ++r)
      st[(r + 8 * hi) * HH + t * 16 + l15] = (_Float16)fmaxf(hf[t][r], 0.f);
  DS_FENCE();

  // ---- a2 = hidden @ at_w2 + at_b2
  fa0 = load_a_frag(st, lane, 0); fa1 = load_a_frag(st, lane, 1);
  v8f attn[4];
#pragma unroll
  for (int nt = 0; nt < 4; ++nt) {
    v8f acc;
    float bias = s_atb2[nt * 16 + l15];
#pragma unroll
    for (int r = 0; r < 8; ++r) acc[r] = bias;
    acc = wmma_f16(fa0, load_b_frag(s_atw2, lane, 0, nt), acc);
    acc = wmma_f16(fa1, load_b_frag(s_atw2, lane, 1, nt), acc);
    attn[nt] = acc;
  }

  // ---- softmax over K (the M dim): 8 regs in-lane + half-wave shfl_xor(16)
#pragma unroll
  for (int t = 0; t < 4; ++t) {
    float m = -3.4e38f;
#pragma unroll
    for (int r = 0; r < 8; ++r) {
      attn[t][r] *= 0.125f;  // 1/sqrt(64)
      m = fmaxf(m, attn[t][r]);
    }
    m = fmaxf(m, __shfl_xor(m, 16));
    float s = 0.f;
#pragma unroll
    for (int r = 0; r < 8; ++r) {
      float e = __expf(attn[t][r] - m);
      attn[t][r] = e;
      s += e;
    }
    s += __shfl_xor(s, 16);
    float inv = 1.f / s;
#pragma unroll
    for (int r = 0; r < 8; ++r) attn[t][r] *= inv;
  }

  // ---- attn output [B,N,K,H]
  float* ab = out_attn + (size_t)p * KK * HH;
#pragma unroll
  for (int t = 0; t < 4; ++t)
#pragma unroll
    for (int r = 0; r < 8; ++r)
      ab[(size_t)(r + 8 * hi) * HH + t * 16 + l15] = attn[t][r];

  // ---- res[h] = sum_k attn[k][h] * (v[k][h] + pos_enc[k][h])
  DS_FENCE();
  stage_gather(st, xvb + (size_t)b * NN * HH, sidx, lane);
  DS_FENCE();

  float* rb = sres[wid];
#pragma unroll
  for (int t = 0; t < 4; ++t) {
    int h = t * 16 + l15;
    float s = 0.f;
#pragma unroll
    for (int r = 0; r < 8; ++r) {
      int k = r + 8 * hi;
      s = fmaf(attn[t][r], (float)st[k * HH + h] + pe[t][r], s);
    }
    s += __shfl_xor(s, 16);
    if (hi == 0) rb[h] = s;
  }
  DS_FENCE();

  // ---- out = res @ Wa + ba + features (residual)
  const float* fr = features + (size_t)p * HH;
  float* orow = out_res + (size_t)p * HH;
#pragma unroll
  for (int cc = 0; cc < 2; ++cc) {
    int c = lane + cc * 32;
    float s = s_ba[c] + fr[c];
#pragma unroll 8
    for (int h2 = 0; h2 < HH; ++h2) s = fmaf(rb[h2], s_Wa[h2 * HH + c], s);
    orow[c] = s;
  }
}

// ---------------------------------------------------------------------------
extern "C" void kernel_launch(void* const* d_in, const int* in_sizes, int n_in,
                              void* d_out, int out_size, void* d_ws,
                              size_t ws_size, hipStream_t stream) {
  (void)in_sizes; (void)n_in; (void)out_size; (void)ws_size;

  const float* xyz      = (const float*)d_in[0];
  const float* features = (const float*)d_in[1];
  const float* Wk       = (const float*)d_in[2];
  const float* bk       = (const float*)d_in[3];
  const float* Wq       = (const float*)d_in[4];
  const float* Wkk      = (const float*)d_in[5];
  const float* Wv       = (const float*)d_in[6];
  const float* pe_w1    = (const float*)d_in[7];
  const float* pe_b1    = (const float*)d_in[8];
  const float* pe_w2    = (const float*)d_in[9];
  const float* pe_b2    = (const float*)d_in[10];
  const float* at_w1    = (const float*)d_in[11];
  const float* at_b1    = (const float*)d_in[12];
  const float* at_w2    = (const float*)d_in[13];
  const float* at_b2    = (const float*)d_in[14];
  const float* Wa       = (const float*)d_in[15];
  const float* ba       = (const float*)d_in[16];
  // d_in[17] = k_nearest (compile-time constant 16 here)

  float* qbuf  = (float*)d_ws;
  float* xkbuf = qbuf + (size_t)PTS * HH;
  float* xvbuf = xkbuf + (size_t)PTS * HH;
  int*   idxb  = (int*)(xvbuf + (size_t)PTS * HH);

  float* out_res  = (float*)d_out;
  float* out_attn = out_res + (size_t)PTS * HH;  // FEAT == 64

  qkv_kernel<<<PTS / (16 * 8), 256, 0, stream>>>(features, Wk, bk, Wq, Wkk, Wv,
                                                 qbuf, xkbuf, xvbuf);
  knn_kernel<<<PTS / 256, 256, 0, stream>>>(xyz, idxb);
  attn_kernel<<<PTS / 8, 256, 0, stream>>>(
      xyz, features, pe_w1, pe_b1, pe_w2, pe_b2, at_w1, at_b1, at_w2, at_b2,
      Wa, ba, qbuf, xkbuf, xvbuf, idxb, out_res, out_attn);
}
